// SS4D_17351667876170
// MI455X (gfx1250) — matrix operand order repaired and checked by reading, not compiled, not run on hardware
//
#include <hip/hip_runtime.h>
#include <hip/hip_bf16.h>

// ---- problem constants (match reference) ----
#define HDIM  512
#define N2    32
#define BDIM  8
#define LDIM  4096
#define TCH   256            // chunk length
#define NC    (LDIM / TCH)   // 16 chunks
#define ODIM  1024           // 2*H
#define KSTEPS (HDIM / 32)   // 16 K-tiles in the GEMM

typedef __attribute__((ext_vector_type(16))) _Float16 v16h;
typedef __attribute__((ext_vector_type(8)))  _Float16 v8h;
typedef __attribute__((ext_vector_type(8)))  float    v8f;

// ---- workspace byte offsets ----
#define WS_W16   0u            // 1024*512 f16        = 1,048,576
#define WS_LAM   1048576u      // 16384 float2        = 131,072
#define WS_CM    1179648u
#define WS_LAMT  1310720u
#define WS_VST   1441792u      // B*H*NC*N2 float2    = 16,777,216
#define WS_SPK   18219008u     // B*L*H f16           = 33,554,432

// Async global->LDS copy of 16 bytes (per active lane), tracked by ASYNCcnt.
// VDST carries the 32-bit LDS byte address: the low 32 bits of a generic
// pointer to __shared__ are exactly the DS-space offset (flat aperture rule).
__device__ __forceinline__ void async_copy_b128(void* lds_dst, const void* gsrc) {
    unsigned lds_addr = (unsigned)(unsigned long long)lds_dst;
    unsigned long long gaddr = (unsigned long long)gsrc;
    asm volatile("global_load_async_to_lds_b128 %0, %1, off"
                 :: "v"(lds_addr), "v"(gaddr) : "memory");
}

template <int N>
__device__ __forceinline__ void wait_asynccnt() {
#if __has_builtin(__builtin_amdgcn_s_wait_asynccnt)
    __builtin_amdgcn_s_wait_asynccnt(N);
#else
    asm volatile("s_wait_asynccnt %0" :: "i"(N) : "memory");
#endif
}

// =====================================================================
// K0: per-(h,n) precompute: lambda = exp(dt*A), Cmod2 = 2*C*(e^{dtA}-1)/A,
//     lambdaT = lambda^TCH
// =====================================================================
__global__ void k_precompute(const float* __restrict__ log_dt,
                             const float* __restrict__ C,
                             const float* __restrict__ log_A_real,
                             const float* __restrict__ A_imag,
                             float2* __restrict__ lam,
                             float2* __restrict__ cm,
                             float2* __restrict__ lamT) {
    int idx = blockIdx.x * 256 + threadIdx.x;       // < H*N2 = 16384
    int h = idx >> 5;
    float dt = __expf(log_dt[h]);
    float Ar = -__expf(log_A_real[idx]);
    float Ai = A_imag[idx];
    float dr = Ar * dt, di = Ai * dt;
    float er = __expf(dr);
    float lr = er * __cosf(di), li = er * __sinf(di);
    lam[idx] = make_float2(lr, li);
    // (e^{dtA}-1)/A
    float xr = lr - 1.0f, xi = li;
    float den = Ar * Ar + Ai * Ai;
    float wr = (xr * Ar + xi * Ai) / den;
    float wi = (xi * Ar - xr * Ai) / den;
    float Cr = C[idx * 2], Ci = C[idx * 2 + 1];
    cm[idx] = make_float2(2.0f * (Cr * wr - Ci * wi),
                          2.0f * (Cr * wi + Ci * wr));
    // lambda^T (accurate libm range reduction; angle up to ~2500 rad)
    float erT = expf(dr * (float)TCH);
    float ang = di * (float)TCH;
    lamT[idx] = make_float2(erT * cosf(ang), erT * sinf(ang));
}

// =====================================================================
// K0b: W (f32, 1024x512 row-major) -> f16
// =====================================================================
__global__ void k_wcast(const float* __restrict__ W, _Float16* __restrict__ w16) {
    int i = blockIdx.x * 256 + threadIdx.x;
    if (i < ODIM * HDIM) w16[i] = (_Float16)W[i];
}

// =====================================================================
// K1: chunk-local end states with zero init.
// One wave per (b,h,chunk); lane = complex mode n (N2 == wave32).
// =====================================================================
__global__ __launch_bounds__(256) void k_chunk_state(
        const float* __restrict__ u,
        const float2* __restrict__ lam,
        float2* __restrict__ vst) {
    int wave = (blockIdx.x << 3) + (threadIdx.x >> 5);
    int lane = threadIdx.x & 31;
    int c =  wave & (NC - 1);
    int h = (wave >> 4) & (HDIM - 1);
    int b =  wave >> 13;
    float2 lm = lam[(h << 5) + lane];
    float sr = 0.0f, si = 0.0f;
    size_t ubase = (((size_t)((b << 9) + h)) << 12) + ((size_t)c << 8);
    for (int blk = 0; blk < TCH / 32; ++blk) {
        float uval = u[ubase + (blk << 5) + lane];
        #pragma unroll
        for (int j = 0; j < 32; ++j) {
            float uv = __shfl(uval, j, 32);
            float nr = lm.x * sr - lm.y * si + uv;
            float ni = lm.x * si + lm.y * sr;
            sr = nr; si = ni;
        }
    }
    vst[(((size_t)((b << 9) + h)) << 9) + (c << 5) + lane] = make_float2(sr, si);
}

// =====================================================================
// K2: in-place exclusive prefix over chunks (16 steps, 1 thread per b,h,n)
// =====================================================================
__global__ void k_prefix(const float2* __restrict__ lamT, float2* __restrict__ vst) {
    int idx = blockIdx.x * 256 + threadIdx.x;      // < B*H*N2 = 131072
    int n =  idx & 31;
    int h = (idx >> 5) & (HDIM - 1);
    int b =  idx >> 14;
    float2 lt = lamT[(h << 5) + n];
    float sr = 0.0f, si = 0.0f;
    size_t base = (((size_t)((b << 9) + h)) << 9) + n;
    #pragma unroll
    for (int c = 0; c < NC; ++c) {
        float2 vc = vst[base + (c << 5)];
        vst[base + (c << 5)] = make_float2(sr, si);   // exclusive prefix
        float nr = lt.x * sr - lt.y * si + vc.x;
        float ni = lt.x * si + lt.y * sr + vc.y;
        sr = nr; si = ni;
    }
}

// =====================================================================
// K3: replay scan with correct initial state; emit spikes in (b,l,h) f16
// (h-contiguous so the GEMM A-side is contiguous in K).
// =====================================================================
__global__ __launch_bounds__(256) void k_scan_spike(
        const float* __restrict__ u,
        const float2* __restrict__ lam,
        const float2* __restrict__ cm,
        const float2* __restrict__ pref,
        const float* __restrict__ Dp,
        const float* __restrict__ ln_vth,
        _Float16* __restrict__ spk) {
    int wave = (blockIdx.x << 3) + (threadIdx.x >> 5);
    int lane = threadIdx.x & 31;
    int c =  wave & (NC - 1);
    int h = (wave >> 4) & (HDIM - 1);
    int b =  wave >> 13;
    float2 lm  = lam[(h << 5) + lane];
    float2 cmv = cm [(h << 5) + lane];
    float2 s0  = pref[(((size_t)((b << 9) + h)) << 9) + (c << 5) + lane];
    float sr = s0.x, si = s0.y;
    float Dh   = Dp[h];
    float ivth = __expf(-ln_vth[h]);
    size_t ubase = (((size_t)((b << 9) + h)) << 12) + ((size_t)c << 8);
    size_t lbase = ((size_t)b << 12) + ((size_t)c << 8);   // l index base
    for (int blk = 0; blk < TCH / 32; ++blk) {
        float uval = u[ubase + (blk << 5) + lane];
        float myspk = 0.0f;
        #pragma unroll
        for (int j = 0; j < 32; ++j) {
            float uv = __shfl(uval, j, 32);
            float nr = lm.x * sr - lm.y * si + uv;
            float ni = lm.x * si + lm.y * sr;
            sr = nr; si = ni;
            float r = cmv.x * sr - cmv.y * si;    // Re(Cmod2 * s), 2x folded in
            r += __shfl_xor(r, 16, 32);
            r += __shfl_xor(r,  8, 32);
            r += __shfl_xor(r,  4, 32);
            r += __shfl_xor(r,  2, 32);
            r += __shfl_xor(r,  1, 32);
            float y = r + Dh * uv;
            float sv = (y * ivth >= 1.0f) ? 1.0f : 0.0f;
            if (j == lane) myspk = sv;            // stage for coalesced-ish store
        }
        spk[(((size_t)(lbase + (blk << 5) + lane)) << 9) + h] = (_Float16)myspk;
    }
}

// =====================================================================
// K4: WMMA GEMM + GLU, double-buffered async global->LDS staging.
//   out_t(l,o) = sum_h spk(b,l,h) * W(o,h), o in [0,1024)
//   res(b,o,l) = (out_t(l,o)+bias[o]) * sigmoid(out_t(l,o+512)+bias[o+512])
// Workgroup: 8 waves -> block 128 l x (64 'a' cols + paired 64 'g' cols).
// Tile i+1's async copies are issued before computing tile i; the wave then
// waits ASYNCcnt<=4 (async loads complete in order) so DMA overlaps WMMA.
// =====================================================================
__global__ __launch_bounds__(256) void k_gemm_glu(
        const _Float16* __restrict__ spk,
        const _Float16* __restrict__ w16,
        const float* __restrict__ bias,
        float* __restrict__ out) {
    __shared__ _Float16 Alds[2][128 * 40];   // 128 l-rows x 32 k (+pad), x2 buf
    __shared__ _Float16 Bals[2][64 * 40];    // 64 o-rows x 32 k (+pad)  (a half)
    __shared__ _Float16 Bgls[2][64 * 40];    //                          (g half)
    __shared__ float    Olds[64 * 132];      // 64 o x 128 l (+pad) transpose

    const int tid = threadIdx.x;
    const int b  = blockIdx.z;
    const int l0 = blockIdx.x * 128;
    const int o0 = blockIdx.y * 64;
    const int wid = tid >> 5;
    const int id  = tid & 31;
    const int m   = id & 15;      // A: M row / B: N col / D: N col
    const int hi  = id >> 4;

    v8f acc_a[4] = { {}, {}, {}, {} };
    v8f acc_g[4] = { {}, {}, {}, {} };

    // 4 async b128 copies per thread per K-tile (2 A segments + Ba + Bg)
    auto issue_tile = [&](int it) {
        const int k0 = it * 32;
        const int bs = it & 1;
        int lin = tid * 2;
        #pragma unroll
        for (int i = 0; i < 2; ++i, ++lin) {
            int row = lin >> 2, seg = lin & 3;
            async_copy_b128(&Alds[bs][row * 40 + seg * 8],
                spk + (((size_t)(b * LDIM + l0 + row)) << 9) + k0 + seg * 8);
        }
        int row = tid >> 2, seg = tid & 3;
        async_copy_b128(&Bals[bs][row * 40 + seg * 8],
            w16 + (size_t)(o0 + row) * HDIM + k0 + seg * 8);
        async_copy_b128(&Bgls[bs][row * 40 + seg * 8],
            w16 + (size_t)(o0 + 512 + row) * HDIM + k0 + seg * 8);
    };

    issue_tile(0);
    for (int it = 0; it < KSTEPS; ++it) {
        const int bs = it & 1;
        if (it + 1 < KSTEPS) {
            issue_tile(it + 1);            // prefill other buffer
            if (it + 2 < KSTEPS)           // and hint L2 two tiles ahead
                __builtin_prefetch(spk + (((size_t)(b * LDIM + l0 + (tid >> 1))) << 9)
                                       + (it + 2) * 32, 0, 1);
            wait_asynccnt<4>();            // this tile done; next still in flight
        } else {
            wait_asynccnt<0>();
        }
        __syncthreads();                   // tile visible to all waves

        // ---- A fragment (16x32 f16): slots 0-7 -> K=hi*8+0..7, 8-15 -> K=16+hi*8
        v8h alo = *(const v8h*)(&Alds[bs][(wid * 16 + m) * 40 + hi * 8]);
        v8h ahi = *(const v8h*)(&Alds[bs][(wid * 16 + m) * 40 + 16 + hi * 8]);
        v16h afrag = __builtin_shufflevector(alo, ahi,
            0,1,2,3,4,5,6,7,8,9,10,11,12,13,14,15);

        #pragma unroll
        for (int ot = 0; ot < 4; ++ot) {
            // B fragment (32x16): lane m = N col, slots 0-7 -> K=hi*8+0..7, 8-15 -> K=16+hi*8
            v8h blo = *(const v8h*)(&Bals[bs][(ot * 16 + m) * 40 + hi * 8]);
            v8h bhi = *(const v8h*)(&Bals[bs][(ot * 16 + m) * 40 + 16 + hi * 8]);
            v16h bfrag = __builtin_shufflevector(blo, bhi,
                0,1,2,3,4,5,6,7,8,9,10,11,12,13,14,15);
            acc_a[ot] = __builtin_amdgcn_wmma_f32_16x16x32_f16(
                false, afrag, false, bfrag, (short)0, acc_a[ot], false, false);

            blo = *(const v8h*)(&Bgls[bs][(ot * 16 + m) * 40 + hi * 8]);
            bhi = *(const v8h*)(&Bgls[bs][(ot * 16 + m) * 40 + 16 + hi * 8]);
            bfrag = __builtin_shufflevector(blo, bhi,
                0,1,2,3,4,5,6,7,8,9,10,11,12,13,14,15);
            acc_g[ot] = __builtin_amdgcn_wmma_f32_16x16x32_f16(
                false, afrag, false, bfrag, (short)0, acc_g[ot], false, false);
        }
        __syncthreads();                   // release this buffer for overwrite
    }

    // ---- GLU in registers, transpose through LDS ----
    #pragma unroll
    for (int ot = 0; ot < 4; ++ot) {
        float ba = bias[o0 + ot * 16 + m];
        float bg = bias[512 + o0 + ot * 16 + m];
        #pragma unroll
        for (int r = 0; r < 8; ++r) {
            float av = acc_a[ot][r] + ba;
            float gv = acc_g[ot][r] + bg;
            float res = av * (1.0f / (1.0f + __expf(-gv)));
            int row_o = ot * 16 + m;             // D: lane = N(o)
            int col_l = wid * 16 + r + hi * 8;   // D: M = vgpr + 8*hi
            Olds[row_o * 132 + col_l] = res;
        }
    }
    __syncthreads();
    #pragma unroll
    for (int i = 0; i < 32; ++i) {
        int idx2 = i * 256 + tid;
        int row = idx2 >> 7, col = idx2 & 127;
        out[((size_t)(b * 512 + o0 + row)) * LDIM + l0 + col] = Olds[row * 132 + col];
    }
}

// =====================================================================
extern "C" void kernel_launch(void* const* d_in, const int* in_sizes, int n_in,
                              void* d_out, int out_size, void* d_ws, size_t ws_size,
                              hipStream_t stream) {
    const float* u          = (const float*)d_in[0];
    const float* log_dt     = (const float*)d_in[1];
    const float* C          = (const float*)d_in[2];
    const float* log_A_real = (const float*)d_in[3];
    const float* A_imag     = (const float*)d_in[4];
    const float* Dp         = (const float*)d_in[5];
    const float* ln_vth     = (const float*)d_in[6];
    const float* W          = (const float*)d_in[7];
    const float* bias       = (const float*)d_in[8];
    float* out = (float*)d_out;

    char* ws = (char*)d_ws;
    _Float16* w16  = (_Float16*)(ws + WS_W16);
    float2*   lam  = (float2*)(ws + WS_LAM);
    float2*   cm   = (float2*)(ws + WS_CM);
    float2*   lamT = (float2*)(ws + WS_LAMT);
    float2*   vst  = (float2*)(ws + WS_VST);
    _Float16* spk  = (_Float16*)(ws + WS_SPK);

    k_precompute<<<HDIM * N2 / 256, 256, 0, stream>>>(log_dt, C, log_A_real,
                                                      A_imag, lam, cm, lamT);
    k_wcast<<<ODIM * HDIM / 256, 256, 0, stream>>>(W, w16);
    k_chunk_state<<<BDIM * HDIM * NC / 8, 256, 0, stream>>>(u, lam, vst);
    k_prefix<<<BDIM * HDIM * N2 / 256, 256, 0, stream>>>(lamT, vst);
    k_scan_spike<<<BDIM * HDIM * NC / 8, 256, 0, stream>>>(u, lam, cm, vst,
                                                           Dp, ln_vth, spk);
    dim3 g(LDIM / 128, 512 / 64, BDIM);
    k_gemm_glu<<<g, 256, 0, stream>>>(spk, w16, bias, out);
}
